// LDPC5GEncoder_54589034332489
// MI455X (gfx1250) — 8-bit matrix operand order repaired and checked by reading, compiled, not run on hardware
//
#include <hip/hip_runtime.h>
#include <stdint.h>

#define Z        384
#define KBITS    8000
#define K_LDPC   8448
#define N_OUT    16000
#define BATCH    2048
#define M_A      1536
#define PB_ROWS  7232

#define OFF_UB8  ((size_t)0)
#define OFF_UT8  ((size_t)17301504)
#define OFF_A8   ((size_t)34603008)
#define OFF_AU   ((size_t)47579136)
#define OFF_PA   ((size_t)50724864)
#define OFF_PB   ((size_t)53870592)

typedef int v8i __attribute__((ext_vector_type(8)));

__global__ void ldpc_pack_kernel(const float* __restrict__ u,
                                 uint8_t* __restrict__ uB8,
                                 uint8_t* __restrict__ uT8) {
    __shared__ uint8_t tile[64][68];
    const int k0 = blockIdx.x * 64;
    const int b0 = blockIdx.y * 64;
    const int tid = threadIdx.x;
    for (int t = tid; t < 4096; t += 256) {
        const int bb = t >> 6, kk = t & 63;
        const int k = k0 + kk, b = b0 + bb;
        uint8_t v = 0;
        if (k < KBITS) v = (uint8_t)(u[(size_t)b * KBITS + k] != 0.0f);
        uB8[(size_t)b * K_LDPC + k] = v;
        tile[kk][bb] = v;
    }
    __syncthreads();
    for (int t = tid; t < 4096; t += 256) {
        const int kk = t >> 6, bb = t & 63;
        uT8[(size_t)(k0 + kk) * BATCH + b0 + bb] = tile[kk][bb];
    }
}

__global__ void ldpc_clear_kernel(uint32_t* __restrict__ p) {
    p[(size_t)blockIdx.x * blockDim.x + threadIdx.x] = 0u;
}

__global__ void ldpc_scatter_kernel(uint8_t* __restrict__ A8,
                                    const int* __restrict__ rows,
                                    const int* __restrict__ cols, int nnz) {
    const int i = blockIdx.x * blockDim.x + threadIdx.x;
    if (i < nnz) A8[(size_t)rows[i] * K_LDPC + cols[i]] = 1;
}

__global__ void ldpc_wmma_au_kernel(const uint8_t* __restrict__ A8,
                                    const uint8_t* __restrict__ uB8,
                                    uint8_t* __restrict__ auT8) {
    const int lane = threadIdx.x;
    const int m0 = blockIdx.x * 16;
    const int n0 = blockIdx.y * 16;
    const int l15 = lane & 15;
    const int half = lane >> 4;
    const uint8_t* arow = A8  + (size_t)(m0 + l15) * K_LDPC;
    const uint8_t* bcol = uB8 + (size_t)(n0 + l15) * K_LDPC;
    v8i acc = {0, 0, 0, 0, 0, 0, 0, 0};
    for (int kk = 0; kk < K_LDPC; kk += 64) {
        v8i af, bf;
#pragma unroll
        for (int v = 0; v < 8; ++v) {
            const int ka = kk + ((v >> 1) << 4) + (half << 3) + ((v & 1) << 2);
            af[v] = *(const int*)(arow + ka);
            const int kb = kk + ((v >> 1) << 4) + (half << 3) + ((v & 1) << 2);
            bf[v] = *(const int*)(bcol + kb);
        }
        acc = __builtin_amdgcn_wmma_i32_16x16x64_iu8(false, af, false, bf, acc,
                                                     false, false);
    }
#pragma unroll
    for (int g = 0; g < 8; ++g) {
        const int m = m0 + g + (half << 3);
        auT8[(size_t)m * BATCH + n0 + l15] = (uint8_t)(acc[g] & 1);
    }
}

__global__ void ldpc_prefix_pa_kernel(const uint8_t* __restrict__ auT8,
                                      uint8_t* __restrict__ paT8) {
    const int t = blockIdx.x * blockDim.x + threadIdx.x;
    const int r = t >> 7;
    const int c = (t & 127) << 4;
    uint4 acc = {0u, 0u, 0u, 0u};
#pragma unroll
    for (int i = 0; i < 4; ++i) {
        const size_t row = (size_t)(i * Z + r) * BATCH + c;
        const uint4 v = *(const uint4*)(auT8 + row);
        acc.x ^= v.x; acc.y ^= v.y; acc.z ^= v.z; acc.w ^= v.w;
        *(uint4*)(paT8 + row) = acc;
    }
}

__device__ __forceinline__ int lb_search(const int* __restrict__ a, int n, int v) {
    int lo = 0, hi = n;
    while (lo < hi) { const int mid = (lo + hi) >> 1; if (a[mid] < v) lo = mid + 1; else hi = mid; }
    return lo;
}

__global__ void ldpc_pb_kernel(const uint8_t* __restrict__ uT8,
                               const uint8_t* __restrict__ paT8,
                               const int* __restrict__ c1r, const int* __restrict__ c1c, int n1,
                               const int* __restrict__ c2r, const int* __restrict__ c2c, int n2,
                               uint8_t* __restrict__ pbT8) {
    const int r = blockIdx.x;
    const int tid = threadIdx.x;
    const int s1 = lb_search(c1r, n1, r), e1 = lb_search(c1r, n1, r + 1);
    const int s2 = lb_search(c2r, n2, r), e2 = lb_search(c2r, n2, r + 1);
    unsigned long long acc = 0ull;
    for (int i = s1; i < e1; ++i)
        acc ^= *(const unsigned long long*)(uT8 + (size_t)c1c[i] * BATCH + tid * 8);
    for (int i = s2; i < e2; ++i)
        acc ^= *(const unsigned long long*)(paT8 + (size_t)c2c[i] * BATCH + tid * 8);
    *(unsigned long long*)(pbT8 + (size_t)r * BATCH + tid * 8) = acc;
}

__global__ void ldpc_out_kernel(const uint8_t* __restrict__ uT8,
                                const uint8_t* __restrict__ paT8,
                                const uint8_t* __restrict__ pbT8,
                                const int* __restrict__ out_int,
                                float* __restrict__ out) {
    __shared__ const uint8_t* rowp[64];
    __shared__ uint8_t tile[64][68];
    const int i0 = blockIdx.x * 64;
    const int b0 = blockIdx.y * 64;
    const int tid = threadIdx.x;
    if (tid < 64) {
        const int j = 768 + out_int[i0 + tid];
        const uint8_t* p;
        if (j < KBITS) p = uT8 + (size_t)j * BATCH;
        else {
            const int jj = j - KBITS;
            p = (jj < M_A) ? (paT8 + (size_t)jj * BATCH)
                           : (pbT8 + (size_t)(jj - M_A) * BATCH);
        }
        rowp[tid] = p;
    }
    __syncthreads();
    for (int t = tid; t < 4096; t += 256) {
        const int ii = t >> 6, bb = t & 63;
        tile[ii][bb] = rowp[ii][b0 + bb];
    }
    __syncthreads();
    for (int t = tid; t < 4096; t += 256) {
        const int ii = t & 63, bb = t >> 6;
        __builtin_nontemporal_store((float)tile[ii][bb],
                                    out + (size_t)(b0 + bb) * N_OUT + i0 + ii);
    }
}

extern "C" void kernel_launch(void* const* d_in, const int* in_sizes, int n_in,
                              void* d_out, int out_size, void* d_ws, size_t ws_size,
                              hipStream_t stream) {
    const float* u      = (const float*)d_in[0];
    const int* a_rows   = (const int*)d_in[1];
    const int* a_cols   = (const int*)d_in[2];
    const int* c1_rows  = (const int*)d_in[5];
    const int* c1_cols  = (const int*)d_in[6];
    const int* c2_rows  = (const int*)d_in[7];
    const int* c2_cols  = (const int*)d_in[8];
    const int* out_int  = (const int*)d_in[9];
    const int nnzA = in_sizes[1];
    const int nnz1 = in_sizes[5];
    const int nnz2 = in_sizes[7];

    uint8_t* ws   = (uint8_t*)d_ws;
    uint8_t* uB8  = ws + OFF_UB8;
    uint8_t* uT8  = ws + OFF_UT8;
    uint8_t* A8   = ws + OFF_A8;
    uint8_t* auT8 = ws + OFF_AU;
    uint8_t* paT8 = ws + OFF_PA;
    uint8_t* pbT8 = ws + OFF_PB;
    float* out = (float*)d_out;

    ldpc_pack_kernel<<<dim3(K_LDPC / 64, BATCH / 64), 256, 0, stream>>>(u, uB8, uT8);
    ldpc_clear_kernel<<<(M_A * K_LDPC / 4) / 256, 256, 0, stream>>>((uint32_t*)A8);
    ldpc_scatter_kernel<<<(nnzA + 255) / 256, 256, 0, stream>>>(A8, a_rows, a_cols, nnzA);
    ldpc_wmma_au_kernel<<<dim3(M_A / 16, BATCH / 16), 32, 0, stream>>>(A8, uB8, auT8);
    ldpc_prefix_pa_kernel<<<(Z * 128) / 256, 256, 0, stream>>>(auT8, paT8);
    ldpc_pb_kernel<<<PB_ROWS, 256, 0, stream>>>(uT8, paT8, c1_rows, c1_cols, nnz1,
                                                c2_rows, c2_cols, nnz2, pbT8);
    ldpc_out_kernel<<<dim3(N_OUT / 64, BATCH / 64), 256, 0, stream>>>(
        uT8, paT8, pbT8, out_int, out);
}
